// QwenGQA_90065464197696
// MI455X (gfx1250) — matrix operand order repaired and checked by reading, not compiled, not run on hardware
//
#include <hip/hip_runtime.h>

// ---------------------------------------------------------------------------
// QwenGQA fused attention layer for MI455X (gfx1250, wave32, WMMA).
// Pipeline:
//   1) cast x / wq|wk|wv (concat) / wo  f32 -> bf16   (workspace)
//   2) qkv = x @ wcat^T     (bf16 WMMA GEMM, 32x64 wave tile, double-buffered)
//   3) RoPE on q,k in-place (f32 VALU)
//   4) v -> vt[b][hkv][d][s]
//   5) flash attention      (async global->LDS double-buffered K/V staging,
//                            WMMA QK^T -> online softmax -> WMMA P^T V)
//   6) out = attn @ wo^T    (f32 out)
// Workspace usage: ~84 MB.
// ---------------------------------------------------------------------------

typedef __bf16 bf16;
typedef __bf16 bf16x8 __attribute__((ext_vector_type(8)));
typedef __bf16 v16bf  __attribute__((ext_vector_type(16)));
typedef float  v8f    __attribute__((ext_vector_type(8)));

#define WMMA_BF16(a, b, c)                                                    \
  __builtin_amdgcn_wmma_f32_16x16x32_bf16(false, (a), false, (b), (short)0,   \
                                          (c), false, false)

// A-matrix (16xK role) fragment: element (m, k) = base[m*ld + k].
// lanes 0-15 -> m=lane, k in {0..7, 16..23}; lanes 16-31 -> m=lane-16,
// k in {8..15, 24..31}. Two contiguous 16B loads.
static __device__ __forceinline__ v16bf frag_a16(const bf16* base, int ld,
                                                 int lane) {
  int m  = lane & 15;
  int kb = (lane < 16) ? 0 : 8;
  const bf16* p = base + (size_t)m * ld + kb;
  bf16x8 lo = *reinterpret_cast<const bf16x8*>(p);
  bf16x8 hi = *reinterpret_cast<const bf16x8*>(p + 16);
  return __builtin_shufflevector(lo, hi, 0, 1, 2, 3, 4, 5, 6, 7, 8, 9, 10, 11,
                                 12, 13, 14, 15);
}

// B-matrix (Kx16 role) fragment: element (k, n) = base[n*ld + k].
// lanes 0-15 -> n=lane, k in {0..7, 16..23}; lanes 16-31 -> n=lane-16,
// k in {8..15, 24..31}.
static __device__ __forceinline__ v16bf frag_b16(const bf16* base, int ld,
                                                 int lane) {
  int n  = lane & 15;
  int kh = (lane < 16) ? 0 : 8;
  const bf16* p = base + (size_t)n * ld + kh;
  bf16x8 lo = *reinterpret_cast<const bf16x8*>(p);
  bf16x8 hi = *reinterpret_cast<const bf16x8*>(p + 16);
  return __builtin_shufflevector(lo, hi, 0, 1, 2, 3, 4, 5, 6, 7, 8, 9, 10, 11,
                                 12, 13, 14, 15);
}

// Raw-pointer variants (per-lane offsets already folded into the pointer).
static __device__ __forceinline__ v16bf lda_raw(const bf16* p) {
  bf16x8 lo = *reinterpret_cast<const bf16x8*>(p);
  bf16x8 hi = *reinterpret_cast<const bf16x8*>(p + 16);
  return __builtin_shufflevector(lo, hi, 0, 1, 2, 3, 4, 5, 6, 7, 8, 9, 10, 11,
                                 12, 13, 14, 15);
}
static __device__ __forceinline__ v16bf ldb_raw(const bf16* p) {
  bf16x8 lo = *reinterpret_cast<const bf16x8*>(p);
  bf16x8 hi = *reinterpret_cast<const bf16x8*>(p + 16);
  return __builtin_shufflevector(lo, hi, 0, 1, 2, 3, 4, 5, 6, 7, 8, 9, 10, 11,
                                 12, 13, 14, 15);
}

// Async copy of 16 bytes per lane, global -> LDS (CDNA5 ASYNCcnt path).
// LDS destination address = low 32 bits of the generic shared pointer
// (ISA 10.2: flat->LDS mapping uses addr[31:0] as wave-relative LDS addr).
static __device__ __forceinline__ void async_g2l_b128(const bf16* g,
                                                      const bf16* l) {
  unsigned loff = (unsigned)(size_t)l;
  asm volatile("global_load_async_to_lds_b128 %0, %1, off"
               :: "v"(loff), "v"(g)
               : "memory");
}
static __device__ __forceinline__ void wait_asynccnt0() {
  asm volatile("s_wait_asynccnt 0x0" ::: "memory");
}

// ------------------------------- cast --------------------------------------
__global__ void cast_f32_bf16(const float* __restrict__ src,
                              bf16* __restrict__ dst, int n) {
  int i = blockIdx.x * blockDim.x + threadIdx.x;
  if (i < n) dst[i] = (bf16)src[i];
}

// ------------------------------- GEMM --------------------------------------
// C[m][n] = sum_k A[m][k] * Bw[n][k].
// Block 256 = 8 waves arranged 2M x 4N; wave tile 32x64 (8 accumulators);
// block tile 64x256. Next k-step's 6 fragments are prefetched into separate
// registers before the current step's 8 wmmas (software double-buffer).
template <bool OUT_F32>
__global__ __launch_bounds__(256) void gemm_bf16_nt(
    const bf16* __restrict__ A, const bf16* __restrict__ Bw,
    void* __restrict__ Cout, int M, int N, int K) {
  int lane = threadIdx.x & 31;
  int w    = threadIdx.x >> 5;
  int wm   = w & 1;
  int wn   = w >> 1;
  int m0   = blockIdx.y * 64 + wm * 32;
  int n0   = blockIdx.x * 256 + wn * 64;

  v8f zero = {0.f, 0.f, 0.f, 0.f, 0.f, 0.f, 0.f, 0.f};
  v8f acc[8];
#pragma unroll
  for (int i = 0; i < 8; ++i) acc[i] = zero;

  const bf16* ap0 = A + (size_t)(m0 + (lane & 15)) * K + ((lane < 16) ? 0 : 8);
  const bf16* ap1 = ap0 + (size_t)16 * K;
  const bf16* bp  = Bw + (size_t)(n0 + (lane & 15)) * K + ((lane < 16) ? 0 : 8);
  const size_t bK = (size_t)16 * K;

  // prologue: k-step 0 fragments
  v16bf a0 = lda_raw(ap0), a1 = lda_raw(ap1);
  v16bf b0 = ldb_raw(bp), b1 = ldb_raw(bp + bK), b2 = ldb_raw(bp + 2 * bK),
        b3 = ldb_raw(bp + 3 * bK);

  for (int k0 = 0; k0 < K; k0 += 32) {
    v16bf ca0 = a0, ca1 = a1, cb0 = b0, cb1 = b1, cb2 = b2, cb3 = b3;
    int k1 = k0 + 32;
    if (k1 < K) {  // uniform branch: prefetch next step into fresh registers
      __builtin_prefetch(ap0 + k1 + 256, 0, 0);
      a0 = lda_raw(ap0 + k1);
      a1 = lda_raw(ap1 + k1);
      b0 = ldb_raw(bp + k1);
      b1 = ldb_raw(bp + bK + k1);
      b2 = ldb_raw(bp + 2 * bK + k1);
      b3 = ldb_raw(bp + 3 * bK + k1);
    }
    acc[0] = WMMA_BF16(ca0, cb0, acc[0]);
    acc[1] = WMMA_BF16(ca0, cb1, acc[1]);
    acc[2] = WMMA_BF16(ca0, cb2, acc[2]);
    acc[3] = WMMA_BF16(ca0, cb3, acc[3]);
    acc[4] = WMMA_BF16(ca1, cb0, acc[4]);
    acc[5] = WMMA_BF16(ca1, cb1, acc[5]);
    acc[6] = WMMA_BF16(ca1, cb2, acc[6]);
    acc[7] = WMMA_BF16(ca1, cb3, acc[7]);
  }

  int mb = (lane < 16) ? 0 : 8;
  int nc = lane & 15;
#pragma unroll
  for (int g = 0; g < 2; ++g) {
#pragma unroll
    for (int t = 0; t < 4; ++t) {
#pragma unroll
      for (int r = 0; r < 8; ++r) {
        size_t off =
            (size_t)(m0 + g * 16 + r + mb) * N + (size_t)(n0 + t * 16 + nc);
        if constexpr (OUT_F32)
          ((float*)Cout)[off] = acc[g * 4 + t][r];
        else
          ((bf16*)Cout)[off] = (bf16)acc[g * 4 + t][r];
      }
    }
  }
}

// ------------------------------- RoPE --------------------------------------
__global__ void rope_kernel(bf16* __restrict__ qkv) {
  const int S = 2048, LD = 3072;
  int idx  = blockIdx.x * blockDim.x + threadIdx.x;  // B*S*20*64 total
  int j    = idx & 63;
  int rest = idx >> 6;
  int h    = rest % 20;
  rest    /= 20;
  int s    = rest & 2047;
  int b    = rest >> 11;
  int col  = (h < 16) ? h * 128 : 2048 + (h - 16) * 128;
  bf16* p  = qkv + ((size_t)b * S + s) * LD + col + j;
  float t1 = (float)p[0];
  float t2 = (float)p[64];
  float ang = (float)s * __powf(10000.f, -(float)j * (1.0f / 64.0f));
  float sn, cs;
  __sincosf(ang, &sn, &cs);
  p[0]  = (bf16)(t1 * cs - t2 * sn);
  p[64] = (bf16)(t2 * cs + t1 * sn);
}

// --------------------------- V transpose -----------------------------------
// vt[((b*4 + hkv)*128 + d)*S + s] = v[b][s][hkv*128 + d]
__global__ void transpose_v(const bf16* __restrict__ qkv,
                            bf16* __restrict__ vt) {
  const int S = 2048, LD = 3072;
  int idx  = blockIdx.x * blockDim.x + threadIdx.x;  // 2*4*128*2048 total
  int s    = idx & 2047;
  int rest = idx >> 11;
  int d    = rest & 127;
  rest   >>= 7;
  int hh   = rest & 3;
  int b    = rest >> 2;
  vt[(((size_t)b * 4 + hh) * 128 + d) * S + s] =
      qkv[((size_t)b * S + s) * LD + 2560 + hh * 128 + d];
}

// --------------------------- flash attention -------------------------------
// Block = 4 waves (128 threads), one (b,h); wave w owns q-tile q0=blk*64+w*16.
// K/V tiles for each 32-kv block are staged cooperatively into LDS with
// global_load_async_to_lds_b128 (double-buffered, ASYNCcnt-tracked); next
// block's staging overlaps the current block's 16 wmmas + softmax.
// All waves run to the block-max causal bound; fully masked blocks are exact
// no-ops in the online softmax (bm=-inf -> alpha=1, p=0).
// Scores computed transposed (St[kv][q] = K x Q^T): softmax over kv is
// within-lane + one shfl_xor(16), and the St C-layout maps 1:1 onto the
// 16x32 bf16 A-fragment for P^T (pure per-lane repack).
#define KSTR 136  // LDS K row stride (elems): 272B = 17*16B, conflict-free
#define VSTR 56   // LDS V row stride (elems): 112B = 7*16B, conflict-free
#define KBUF_ELEMS (32 * KSTR)    // 4352 elems / 8704 B
#define VBUF_ELEMS (128 * VSTR)   // 7168 elems / 14336 B

__global__ __launch_bounds__(128) void attn_fa(const bf16* __restrict__ qkv,
                                               const bf16* __restrict__ vt,
                                               bf16* __restrict__ attn) {
  const int S = 2048, LD = 3072, HD = 128;
  const float scale = 0.088388347648318447f;  // 1/sqrt(128)
  const float NEG_INF = -__builtin_inff();

  __shared__ __align__(16) bf16 smem[2 * KBUF_ELEMS + 2 * VBUF_ELEMS];

  int tid  = threadIdx.x;
  int lane = tid & 31;
  int wave = tid >> 5;
  int q0   = (blockIdx.x * 4 + wave) * 16;
  int h    = blockIdx.y;
  int b    = blockIdx.z;
  int hk   = h >> 2;  // GQA: 4 query heads per kv head

  const bf16* qptr = qkv + ((size_t)b * S + q0) * LD + h * HD;
  const bf16* kptr = qkv + (size_t)b * S * LD + 2048 + hk * HD;
  const bf16* vptr = vt + ((size_t)b * 4 + hk) * (size_t)HD * S;

  // Q^T B-fragments for the 4 hd-chunks of 32; reused across all kv blocks.
  v16bf qf[4];
#pragma unroll
  for (int c = 0; c < 4; ++c) qf[c] = frag_b16(qptr + c * 32, LD, lane);

  v8f zero = {0.f, 0.f, 0.f, 0.f, 0.f, 0.f, 0.f, 0.f};
  v8f o[8];
#pragma unroll
  for (int t = 0; t < 8; ++t) o[t] = zero;

  float m_run = NEG_INF;
  float l_run = 0.f;
  int qi = q0 + (lane & 15);     // q column owned by this lane (St layout)
  int rb = (lane < 16) ? 0 : 8;  // kv row base of this lane half
  int srcadd = rb;               // shfl source offset for O-row broadcast

  // cooperative async staging of one 32-kv block into LDS buffer nb
  auto stage = [&](int kv0, int nb) {
    bf16* kdst = smem + nb * KBUF_ELEMS;
    bf16* vdst = smem + 2 * KBUF_ELEMS + nb * VBUF_ELEMS;
    // K tile: 32 rows x 128 bf16 (16 x 16B chunks per row), 512 chunks
#pragma unroll
    for (int i = 0; i < 4; ++i) {
      int chunk = tid + i * 128;
      int row   = chunk >> 4;
      int c8    = chunk & 15;
      async_g2l_b128(kptr + (size_t)(kv0 + row) * LD + c8 * 8,
                     kdst + row * KSTR + c8 * 8);
    }
    // V^T tile: 128 d-rows x 32 kv (4 x 16B chunks per row), 512 chunks
#pragma unroll
    for (int i = 0; i < 4; ++i) {
      int chunk = tid + i * 128;
      int row   = chunk >> 2;
      int c8    = chunk & 3;
      async_g2l_b128(vptr + (size_t)row * S + kv0 + c8 * 8,
                     vdst + row * VSTR + c8 * 8);
    }
  };

  // number of 32-kv blocks up to the block-max causal bound (q = blk*64+63)
  int nblk = blockIdx.x * 2 + 2;

  stage(0, 0);
  wait_asynccnt0();
  __syncthreads();

  for (int ib = 0; ib < nblk; ++ib) {
    int nb  = ib & 1;
    int kv0 = ib * 32;
    if (ib + 1 < nblk) stage((ib + 1) * 32, 1 - nb);  // overlaps compute

    const bf16* kb = smem + nb * KBUF_ELEMS;
    const bf16* vb = smem + 2 * KBUF_ELEMS + nb * VBUF_ELEMS;

    // fragments from LDS (ds_load_b128, low latency)
    v16bf kf[8];
#pragma unroll
    for (int c = 0; c < 4; ++c) {
      kf[c]     = frag_a16(kb + c * 32, KSTR, lane);
      kf[c + 4] = frag_a16(kb + 16 * KSTR + c * 32, KSTR, lane);
    }
    v16bf vf[8];
#pragma unroll
    for (int t = 0; t < 8; ++t)
      vf[t] = frag_b16(vb + (size_t)(t * 16) * VSTR, VSTR, lane);

    // scores (transposed): St = K x Q^T over the 4 hd chunks
    v8f s1 = zero, s2 = zero;
#pragma unroll
    for (int c = 0; c < 4; ++c) {
      s1 = WMMA_BF16(kf[c], qf[c], s1);
      s2 = WMMA_BF16(kf[c + 4], qf[c], s2);
    }

    // causal mask + block max (per q column)
    float p1[8], p2[8];
    float bm = NEG_INF;
#pragma unroll
    for (int r = 0; r < 8; ++r) {
      int kv = kv0 + rb + r;
      p1[r] = (kv <= qi) ? s1[r] * scale : NEG_INF;
      p2[r] = (kv + 16 <= qi) ? s2[r] * scale : NEG_INF;
      bm = fmaxf(bm, fmaxf(p1[r], p2[r]));
    }
    bm = fmaxf(bm, __shfl_xor(bm, 16, 32));

    float m_new = fmaxf(m_run, bm);
    float alpha = __expf(m_run - m_new);  // first block: exp(-inf) = 0
    float rsum = 0.f;
#pragma unroll
    for (int r = 0; r < 8; ++r) {
      p1[r] = __expf(p1[r] - m_new);
      p2[r] = __expf(p2[r] - m_new);
      rsum += p1[r] + p2[r];
    }
    rsum += __shfl_xor(rsum, 16, 32);
    l_run = l_run * alpha + rsum;
    m_run = m_new;

    // Rescale O accumulators: O rows are q = r + rb; alpha lives at lane q.
#pragma unroll
    for (int r = 0; r < 8; ++r) {
      float ar = __shfl(alpha, r + srcadd, 32);
#pragma unroll
      for (int t = 0; t < 8; ++t) o[t][r] *= ar;
    }

    // P^T A-fragment: St C-layout == A-fragment layout (per-lane repack)
    v16bf pa;
#pragma unroll
    for (int e = 0; e < 8; ++e) {
      pa[e]     = (bf16)p1[e];
      pa[e + 8] = (bf16)p2[e];
    }

#pragma unroll
    for (int t = 0; t < 8; ++t) o[t] = WMMA_BF16(pa, vf[t], o[t]);

    // next buffer ready + everyone done reading current buffer
    wait_asynccnt0();
    __syncthreads();
  }

  // normalize by l and store bf16 into attn[b][s][h*128 + d]
#pragma unroll
  for (int r = 0; r < 8; ++r) {
    float lr  = __shfl(l_run, r + srcadd, 32);
    float inv = 1.f / lr;
    int srow  = q0 + r + rb;
    bf16* orow = attn + ((size_t)b * S + srow) * 2048 + h * HD + (lane & 15);
#pragma unroll
    for (int t = 0; t < 8; ++t) orow[t * 16] = (bf16)(o[t][r] * inv);
  }
}

// ------------------------------- launch ------------------------------------
extern "C" void kernel_launch(void* const* d_in, const int* in_sizes, int n_in,
                              void* d_out, int out_size, void* d_ws,
                              size_t ws_size, hipStream_t stream) {
  (void)in_sizes; (void)n_in; (void)out_size; (void)ws_size;
  const float* x  = (const float*)d_in[0];
  const float* wq = (const float*)d_in[1];
  const float* wk = (const float*)d_in[2];
  const float* wv = (const float*)d_in[3];
  const float* wo = (const float*)d_in[4];

  char* ws   = (char*)d_ws;
  bf16* x16  = (bf16*)(ws + 0);          // 16,777,216 B
  bf16* wcat = (bf16*)(ws + 16777216);   // 12,582,912 B  (wq|wk|wv rows)
  bf16* wo16 = (bf16*)(ws + 29360128);   //  8,388,608 B
  bf16* qkv  = (bf16*)(ws + 37748736);   // 25,165,824 B  [4096][3072]
  bf16* vt   = (bf16*)(ws + 62914560);   //  4,194,304 B  [2][4][128][2048]
  bf16* attn = (bf16*)(ws + 67108864);   // 16,777,216 B  [4096][2048]

  auto cast = [&](const float* s, bf16* d, int n) {
    cast_f32_bf16<<<(n + 255) / 256, 256, 0, stream>>>(s, d, n);
  };
  cast(x, x16, 8388608);
  cast(wq, wcat, 4194304);
  cast(wk, wcat + 4194304, 1048576);
  cast(wv, wcat + 5242880, 1048576);
  cast(wo, wo16, 4194304);

  // qkv = x @ [wq;wk;wv]^T   (M=4096, N=3072, K=2048), block tile 64x256
  gemm_bf16_nt<false><<<dim3(12, 64), 256, 0, stream>>>(x16, wcat, qkv, 4096,
                                                        3072, 2048);
  // RoPE on q,k (2*2048*20*64 threads)
  rope_kernel<<<20480, 256, 0, stream>>>(qkv);
  // v -> vt (2*4*128*2048 threads)
  transpose_v<<<8192, 256, 0, stream>>>(qkv, vt);
  // flash attention: grid (S/64, HQ, B), 4 waves per block
  attn_fa<<<dim3(32, 16, 2), 128, 0, stream>>>(qkv, vt, attn);
  // out = attn @ wo^T  (f32 out), M=4096, N=2048, K=2048, block tile 64x256
  gemm_bf16_nt<true><<<dim3(8, 64), 256, 0, stream>>>(attn, wo16, d_out, 4096,
                                                      2048, 2048);
}
